// LinearGML2_62405874811525
// MI455X (gfx1250) — matrix operand order repaired and checked by reading, not compiled, not run on hardware
//
#include <hip/hip_runtime.h>
#include <hip/hip_bf16.h>
#include <math.h>

// ---- CDNA5 WMMA operand types --------------------------------------------
typedef __attribute__((ext_vector_type(16))) __bf16 v16bf;
typedef __attribute__((ext_vector_type(8)))  __bf16 bf8;
typedef __attribute__((ext_vector_type(4)))  __bf16 bf4;
typedef __attribute__((ext_vector_type(8)))  float  v8f;
typedef __attribute__((ext_vector_type(4)))  float  f4;

#define D_FEAT 512
#define K_CLS  256
#define BATCH  1024
#define N_LOWER (D_FEAT * (D_FEAT - 1) / 2)   // 130816, np.tril_indices(d,-1) packing

#define BT   128   // batch rows per workgroup (8 waves x 16-row A tiles)
#define KC   64    // contraction chunk
#define EE   64    // output-column chunk (4 N tiles -> 4 C accumulators/wave)
#define DSTR 72    // padded LDS row stride for sD  (144B rows, 16B aligned)
#define LSTR 72    // padded LDS row stride for sLT (144B rows, 16B aligned)

union ABOp { v16bf v; bf8 h[2]; };

__global__ __launch_bounds__(256, 1)
void gml2_wmma_kernel(const float* __restrict__ X,      // [1024, 512]
                      const float* __restrict__ W,      // [256, 1, 512]
                      const float* __restrict__ Ldiag,  // [256, 512]
                      const float* __restrict__ Llow,   // [256, 130816]
                      float* __restrict__ out)          // [1024, 256]
{
  __shared__ __bf16 sD [BT * DSTR];   // bf16(X - w_k) chunk, A-operand friendly
  __shared__ __bf16 sLT[EE * LSTR];   // L_k tile transposed: [n][kr] (column-major of L)
  __shared__ float  sW [D_FEAT];      // w_k row, f32

  const int k    = blockIdx.x >> 3;          // class; consecutive blocks share k (L2 reuse)
  const int b0   = (blockIdx.x & 7) * BT;    // batch tile origin
  const int tid  = threadIdx.x;
  const int wave = tid >> 5;
  const int lane = tid & 31;
  const int lh   = lane >> 4;                // half-wave select
  const int lm   = lane & 15;
  const int rowA = wave * 16 + lm;           // this lane's A-matrix row (M)

  for (int i = tid; i < D_FEAT; i += 256) sW[i] = W[(size_t)k * D_FEAT + i];

  float acc[8] = {0.f, 0.f, 0.f, 0.f, 0.f, 0.f, 0.f, 0.f};

  for (int ee = 0; ee < D_FEAT; ee += EE) {            // output-column chunk (4 N tiles)
    v8f c0 = {0.f,0.f,0.f,0.f,0.f,0.f,0.f,0.f};
    v8f c1 = c0, c2 = c0, c3 = c0;

    for (int dd = ee; dd < D_FEAT; dd += KC) {         // triangular: L[d,e]=0 for d<e
      __syncthreads();

      // ---- stage D = X - w_k chunk (float4 load, packed bf16 store) -------
      for (int p = tid; p < BT * (KC / 4); p += 256) { // 2048 packets, 8/thread
        const int r  = p >> 4;
        const int c4 = (p & 15) * 4;
        const f4 x = *(const f4*)&X[(size_t)(b0 + r) * D_FEAT + dd + c4];
        const f4 w = *(const f4*)&sW[dd + c4];
        bf4 o;
        o.x = (__bf16)(x.x - w.x);
        o.y = (__bf16)(x.y - w.y);
        o.z = (__bf16)(x.z - w.z);
        o.w = (__bf16)(x.w - w.w);
        *(bf4*)&sD[r * DSTR + c4] = o;                 // ds_store_b64
      }
      // prefetch next X chunk (global_prefetch_b8)
      if (dd + KC < D_FEAT)
        __builtin_prefetch(X + (size_t)(b0 + (tid & 127)) * D_FEAT + dd + KC, 0, 0);

      // ---- stage L_k tile transposed: sLT[n*LSTR+kr] = L_k[dd+kr, ee+n] ---
      for (int i = tid; i < EE * KC; i += 256) {       // 4096 elems, 16/thread
        const int n  = i >> 6;
        const int kr = i & (KC - 1);
        const int d  = dd + kr;
        const int e  = ee + n;
        float v;
        if (e > d) {
          v = 0.f;                                     // above diagonal
        } else if (e == d) {
          const float g = Ldiag[(size_t)k * D_FEAT + d];
          v = g * g;                                   // diag = L_diags^2
        } else {
          v = Llow[(size_t)k * N_LOWER + ((size_t)d * (d - 1)) / 2 + e];
        }
        sLT[n * LSTR + kr] = (__bf16)v;
      }
      __syncthreads();

      // ---- 8 bf16 WMMAs: one A tile x 4 N tiles, two K halves -------------
      ABOp a, b;
      #pragma unroll
      for (int s = 0; s < 2; ++s) {
        const int ka = s * 32;
        // A k order used here (contiguous, not ISA table order): lane<16 -> K ka+0..15; lane>=16 -> K ka+16..31 (matches B)
        a.h[0] = *(const bf8*)&sD[rowA * DSTR + ka + 16 * lh];
        a.h[1] = *(const bf8*)&sD[rowA * DSTR + ka + 16 * lh + 8];
        const int kr0 = ka + 16 * lh;
        // B layout: lane holds column N = (t*16 + lm), K ascending, halves at 16
        b.h[0] = *(const bf8*)&sLT[( 0 + lm) * LSTR + kr0];
        b.h[1] = *(const bf8*)&sLT[( 0 + lm) * LSTR + kr0 + 8];
        c0 = __builtin_amdgcn_wmma_f32_16x16x32_bf16(false, a.v, false, b.v,
                                                     (short)0, c0, false, false);
        b.h[0] = *(const bf8*)&sLT[(16 + lm) * LSTR + kr0];
        b.h[1] = *(const bf8*)&sLT[(16 + lm) * LSTR + kr0 + 8];
        c1 = __builtin_amdgcn_wmma_f32_16x16x32_bf16(false, a.v, false, b.v,
                                                     (short)0, c1, false, false);
        b.h[0] = *(const bf8*)&sLT[(32 + lm) * LSTR + kr0];
        b.h[1] = *(const bf8*)&sLT[(32 + lm) * LSTR + kr0 + 8];
        c2 = __builtin_amdgcn_wmma_f32_16x16x32_bf16(false, a.v, false, b.v,
                                                     (short)0, c2, false, false);
        b.h[0] = *(const bf8*)&sLT[(48 + lm) * LSTR + kr0];
        b.h[1] = *(const bf8*)&sLT[(48 + lm) * LSTR + kr0 + 8];
        c3 = __builtin_amdgcn_wmma_f32_16x16x32_bf16(false, a.v, false, b.v,
                                                     (short)0, c3, false, false);
      }
    }

    // ---- squared-norm accumulation over this 64-wide N chunk --------------
    #pragma unroll
    for (int r = 0; r < 8; ++r) {
      acc[r] += c0[r]*c0[r] + c1[r]*c1[r] + c2[r]*c2[r] + c3[r]*c3[r];
    }
  }

  // ---- reduce over N (16 lanes per half-wave share a row), sqrt, store ----
  #pragma unroll
  for (int r = 0; r < 8; ++r) {
    float v = acc[r];
    v += __shfl_xor(v, 1, 32);
    v += __shfl_xor(v, 2, 32);
    v += __shfl_xor(v, 4, 32);
    v += __shfl_xor(v, 8, 32);   // stays within each 16-lane half
    if (lm == 0) {
      const int row = b0 + wave * 16 + r + 8 * lh;   // C layout: M = r + 8*(lane>=16)
      out[(size_t)row * K_CLS + k] = sqrtf(v);
    }
  }
}

extern "C" void kernel_launch(void* const* d_in, const int* in_sizes, int n_in,
                              void* d_out, int out_size, void* d_ws, size_t ws_size,
                              hipStream_t stream) {
  (void)in_sizes; (void)n_in; (void)d_ws; (void)ws_size; (void)out_size;
  const float* X  = (const float*)d_in[0];   // inputs  [1024, 512]
  const float* W  = (const float*)d_in[1];   // weight  [256, 1, 512]
  const float* Ld = (const float*)d_in[2];   // L_diags [256, 512]
  const float* Ll = (const float*)d_in[3];   // L_lower [256, 130816]
  float* out = (float*)d_out;                // [1024, 256]

  const int grid = K_CLS * (BATCH / BT);     // 256 * 8 = 2048 workgroups
  gml2_wmma_kernel<<<grid, 256, 0, stream>>>(X, W, Ld, Ll, out);
}
